// GatingSASmambaRec_30399778521709
// MI455X (gfx1250) — matrix operand order repaired and checked by reading, not compiled, not run on hardware
//
#include <hip/hip_runtime.h>
#include <hip/hip_bf16.h>
#include <math.h>

typedef _Float16 h16;
typedef __attribute__((ext_vector_type(8)))  _Float16 v8h;
typedef __attribute__((ext_vector_type(16))) _Float16 v16h;
typedef __attribute__((ext_vector_type(8)))  float    v8f;

#define Bsz 64
#define Lseq 200
#define Hdim 256
#define DIc 512
#define Nst 32
#define DRr 16
#define NHEADS 4
#define DH 64
#define Lp 208        // 200 padded to 13 WMMA tiles (M/N padding)
#define Lpk 224       // 200 padded to 7 K-steps of 32 (K padding)
#define QKVW 768
#define QKV_PAD_ROWS 16

__device__ __forceinline__ v16h cat8(v8h lo, v8h hi) {
  return __builtin_shufflevector(lo, hi, 0,1,2,3,4,5,6,7,8,9,10,11,12,13,14,15);
}

// ---------------------------------------------------------------- utilities
__global__ void cvt_f16_kernel(const float* __restrict__ in, h16* __restrict__ out, int n) {
  int i = blockIdx.x * blockDim.x + threadIdx.x;
  if (i < n) out[i] = (h16)in[i];
}

__global__ void zfill_h16_kernel(h16* __restrict__ p, int n) {
  int i = blockIdx.x * blockDim.x + threadIdx.x;
  if (i < n) p[i] = (h16)0.f;
}

// dt_w [512,16] -> zero-padded f16 [512,32] so the dt GEMM can run K=32 guard-free
__global__ void dtw_pad_kernel(const float* __restrict__ w, h16* __restrict__ o) {
  int i = blockIdx.x * blockDim.x + threadIdx.x;
  if (i >= DIc * 32) return;
  int k = i & 31, n = i >> 5;
  o[i] = (k < DRr) ? (h16)w[n * DRr + k] : (h16)0.f;
}

// seqs[b,l,h] = item_emb[log[b,l],h]*sqrt(H) + pos_emb[l,h]
__global__ void embed_kernel(const int* __restrict__ ids, const float* __restrict__ item_emb,
                             const float* __restrict__ pos_emb, float* __restrict__ seqs,
                             h16* __restrict__ seqs_h) {
  int i = blockIdx.x * blockDim.x + threadIdx.x;
  if (i >= Bsz * Lseq * Hdim) return;
  int h  = i & (Hdim - 1);
  int bl = i >> 8;            // Hdim == 256
  int l  = bl % Lseq;
  int it = ids[bl];
  float v = item_emb[(size_t)it * Hdim + h] * 16.0f + pos_emb[l * Hdim + h];
  seqs[i] = v; seqs_h[i] = (h16)v;
}

// per-sample hard gate: argmax of mean_L(seqs) @ gate_w.T + gate_b
__global__ void gate_kernel(const float* __restrict__ seqs, const float* __restrict__ gw,
                            const float* __restrict__ gb, int* __restrict__ gate) {
  __shared__ float red[Hdim];
  int b = blockIdx.x, t = threadIdx.x;
  float s = 0.f;
  for (int l = 0; l < Lseq; ++l) s += seqs[((size_t)(b * Lseq + l)) * Hdim + t];
  float m = s * (1.0f / Lseq);
  red[t] = m * gw[t]; __syncthreads();
  for (int st = 128; st > 0; st >>= 1) { if (t < st) red[t] += red[t + st]; __syncthreads(); }
  float l0 = red[0] + gb[0]; __syncthreads();
  red[t] = m * gw[Hdim + t]; __syncthreads();
  for (int st = 128; st > 0; st >>= 1) { if (t < st) red[t] += red[t + st]; __syncthreads(); }
  if (t == 0) gate[b] = ((red[0] + gb[1]) > l0) ? 1 : 0;
}

// ---------------------------------------------------------------- generic WMMA GEMM
// C[M,N] = act(A[M,K] @ W[N,K]^T + bias). Requires K%32==0, M%128==0, N%16==0.
// 32x32 register tile per wave (2x2 WMMA tiles, fragments reused twice -> 2 b128
// loads per WMMA). N tile is per-block so the half-tile predicate is a scalar
// branch hoisted out of the K loop (no EXEC masking around WMMA). The 4 waves of
// a block split M and share B tiles (L0/L2 locality).
__global__ void gemm_wmma_kernel(const h16* __restrict__ A, int lda,
                                 const h16* __restrict__ W, int ldw,
                                 const float* __restrict__ bias,
                                 float* __restrict__ C, h16* __restrict__ Ch, int ldc,
                                 int N, int K, int act) {
  int wave = threadIdx.x >> 5;
  int lane = threadIdx.x & 31;
  int g = lane >> 4;                       // lane group 0/1
  int n0 = blockIdx.x * 32;                // block-uniform 32-wide N tile
  int m0 = (blockIdx.y * 4 + wave) * 32;   // per-wave 32-tall M tile
  bool hasN2 = (n0 + 16) < N;              // block-uniform -> scalar branch
  // A lane base: row m0+(lane&15), chunk [16g..16g+15] per k-step
  const h16* arow0 = A + (size_t)(m0 + (lane & 15)) * lda + (g << 4);
  const h16* arow1 = arow0 + (size_t)16 * lda;
  // B lane base: col n0+(lane&15), chunk [16g..16g+15] per k-step
  const h16* bcol0 = W + (size_t)(n0 + (lane & 15)) * ldw + (g << 4);
  const h16* bcol1 = bcol0 + (size_t)16 * ldw;
  v8f c00 = {}, c01 = {}, c10 = {}, c11 = {};
  if (hasN2) {
    for (int k0 = 0; k0 < K; k0 += 32) {
      v16h a0 = cat8(*(const v8h*)(arow0 + k0), *(const v8h*)(arow0 + k0 + 8));
      v16h a1 = cat8(*(const v8h*)(arow1 + k0), *(const v8h*)(arow1 + k0 + 8));
      v16h b0 = cat8(*(const v8h*)(bcol0 + k0), *(const v8h*)(bcol0 + k0 + 8));
      v16h b1 = cat8(*(const v8h*)(bcol1 + k0), *(const v8h*)(bcol1 + k0 + 8));
      c00 = __builtin_amdgcn_wmma_f32_16x16x32_f16(false, a0, false, b0, (short)0, c00, false, false);
      c10 = __builtin_amdgcn_wmma_f32_16x16x32_f16(false, a1, false, b0, (short)0, c10, false, false);
      c01 = __builtin_amdgcn_wmma_f32_16x16x32_f16(false, a0, false, b1, (short)0, c01, false, false);
      c11 = __builtin_amdgcn_wmma_f32_16x16x32_f16(false, a1, false, b1, (short)0, c11, false, false);
    }
  } else {
    for (int k0 = 0; k0 < K; k0 += 32) {
      v16h a0 = cat8(*(const v8h*)(arow0 + k0), *(const v8h*)(arow0 + k0 + 8));
      v16h a1 = cat8(*(const v8h*)(arow1 + k0), *(const v8h*)(arow1 + k0 + 8));
      v16h b0 = cat8(*(const v8h*)(bcol0 + k0), *(const v8h*)(bcol0 + k0 + 8));
      c00 = __builtin_amdgcn_wmma_f32_16x16x32_f16(false, a0, false, b0, (short)0, c00, false, false);
      c10 = __builtin_amdgcn_wmma_f32_16x16x32_f16(false, a1, false, b0, (short)0, c10, false, false);
    }
  }
  auto store_tile = [&](v8f c, int mbase, int nbase) {
    int n = nbase + (lane & 15);
    float bv = bias ? bias[n] : 0.f;
    #pragma unroll
    for (int v = 0; v < 8; ++v) {
      int m = mbase + v + (g << 3);        // C layout: VGPR v <-> M = v + 8*lanegrp
      float x = c[v] + bv;
      if      (act == 1) x = x / (1.f + __expf(-x));
      else if (act == 2) x = fmaxf(x, 0.f);
      else if (act == 3) x = (x > 20.f) ? x : log1pf(__expf(x));
      if (C)  C [(size_t)m * ldc + n] = x;
      if (Ch) Ch[(size_t)m * ldc + n] = (h16)x;
    }
  };
  store_tile(c00, m0,      n0);
  store_tile(c10, m0 + 16, n0);
  if (hasN2) {
    store_tile(c01, m0,      n0 + 16);
    store_tile(c11, m0 + 16, n0 + 16);
  }
}

// ---------------------------------------------------------------- mamba pieces
__global__ void conv_silu_kernel(const float* __restrict__ xz, const float* __restrict__ cw,
                                 const float* __restrict__ cb, float* __restrict__ xc,
                                 h16* __restrict__ xch) {
  int i = blockIdx.x * blockDim.x + threadIdx.x;
  if (i >= Bsz * Lseq * DIc) return;
  int d = i % DIc; int bl = i / DIc; int l = bl % Lseq; int b = bl / Lseq;
  float acc = cb[d];
  #pragma unroll
  for (int j = 0; j < 4; ++j) {
    int ls = l - 3 + j;
    if (ls >= 0) acc += cw[d * 4 + j] * xz[((size_t)(b * Lseq + ls)) * 1024 + d];
  }
  float v = acc / (1.f + __expf(-acc));
  xc[i] = v; xch[i] = (h16)v;
}

// selective scan: one thread per (b,d), h[32] state in registers, fused D-skip + silu(z) gate
__global__ void scan_kernel(const float* __restrict__ delta, const float* __restrict__ dbl,
                            const float* __restrict__ xc, const float* __restrict__ xz,
                            const float* __restrict__ A_log, const float* __restrict__ D_skip,
                            float* __restrict__ yg, h16* __restrict__ ygh) {
  int idx = blockIdx.x * blockDim.x + threadIdx.x;
  if (idx >= Bsz * DIc) return;
  int b = idx / DIc, d = idx % DIc;
  float a[Nst], h[Nst];
  #pragma unroll
  for (int n = 0; n < Nst; ++n) { a[n] = -__expf(A_log[d * Nst + n]); h[n] = 0.f; }
  float Dd = D_skip[d];
  for (int l = 0; l < Lseq; ++l) {
    size_t r = (size_t)(b * Lseq + l);
    float dt = delta[r * DIc + d];
    float u  = xc[r * DIc + d];
    const float* Bm = dbl + r * 80 + DRr;
    const float* Cm = dbl + r * 80 + DRr + Nst;
    float du = dt * u, y = 0.f;
    #pragma unroll
    for (int n = 0; n < Nst; ++n) {
      h[n] = __expf(dt * a[n]) * h[n] + du * Bm[n];
      y += h[n] * Cm[n];
    }
    float z = xz[r * 1024 + DIc + d];
    float o = (y + u * Dd) * (z / (1.f + __expf(-z)));
    yg[r * DIc + d] = o; ygh[r * DIc + d] = (h16)o;
  }
}

// ---------------------------------------------------------------- attention (WMMA, batched b*heads)
// qkv has QKV_PAD_ROWS zeroed pad rows past B*L, so the last batch's rows m,n in [200,208) read zeros.
__global__ void attn_scores_kernel(const h16* __restrict__ qkv, float* __restrict__ S) {
  int lane = threadIdx.x & 31;
  int g = lane >> 4;
  int nt = blockIdx.x, mt = blockIdx.y, bh = blockIdx.z;
  int b = bh >> 2, hd = bh & 3;
  const h16* arow = qkv + ((size_t)(b * Lseq + mt * 16 + (lane & 15))) * QKVW
                        + hd * DH + (g << 4);
  const h16* bcol = qkv + ((size_t)(b * Lseq + nt * 16 + (lane & 15))) * QKVW
                        + Hdim + hd * DH + (g << 4);
  v8f c = {};
  #pragma unroll
  for (int k0 = 0; k0 < DH; k0 += 32) {
    v16h a = cat8(*(const v8h*)(arow + k0), *(const v8h*)(arow + k0 + 8));
    v16h bf = cat8(*(const v8h*)(bcol + k0), *(const v8h*)(bcol + k0 + 8));
    c = __builtin_amdgcn_wmma_f32_16x16x32_f16(false, a, false, bf, (short)0, c, false, false);
  }
  float* sr = S + (size_t)bh * Lp * Lp;
  int n = nt * 16 + (lane & 15);
  #pragma unroll
  for (int v = 0; v < 8; ++v) {
    int m = mt * 16 + v + (g << 3);
    sr[(size_t)m * Lp + n] = c[v] * 0.125f;     // 1/sqrt(64)
  }
}

// softmax row (l<200); writes P with row stride Lpk=224, zero in cols [200,224)
__global__ void softmax_kernel(const float* __restrict__ S, h16* __restrict__ P) {
  __shared__ float red[256];
  int row = blockIdx.x;                     // bh*Lseq + l
  int bh = row / Lseq, l = row % Lseq;
  const float* sr = S + ((size_t)bh * Lp + l) * Lp;
  int t = threadIdx.x;
  float x = (t < Lseq) ? sr[t] : -3.0e38f;
  red[t] = x; __syncthreads();
  for (int st = 128; st > 0; st >>= 1) { if (t < st) red[t] = fmaxf(red[t], red[t + st]); __syncthreads(); }
  float mx = red[0]; __syncthreads();
  float e = (t < Lseq) ? __expf(x - mx) : 0.f;
  red[t] = e; __syncthreads();
  for (int st = 128; st > 0; st >>= 1) { if (t < st) red[t] += red[t + st]; __syncthreads(); }
  float inv = 1.f / red[0];
  h16* pr = P + ((size_t)bh * Lp + l) * Lpk;
  if (t < Lpk) pr[t] = (t < Lseq) ? (h16)(e * inv) : (h16)0.f;
}

// V transposed per (bh): vt[bh][n][kk], kk zero-padded to 224 -> contiguous B fragments
__global__ void vtrans_kernel(const h16* __restrict__ qkv, h16* __restrict__ vt) {
  int i = blockIdx.x * blockDim.x + threadIdx.x;
  if (i >= Bsz * NHEADS * DH * Lpk) return;
  int kk = i % Lpk; int rest = i / Lpk; int n = rest % DH; int bh = rest / DH;
  int b = bh >> 2, hd = bh & 3;
  h16 v = (h16)0.f;
  if (kk < Lseq) v = qkv[((size_t)(b * Lseq + kk)) * QKVW + 2 * Hdim + hd * DH + n];
  vt[i] = v;
}

// ctx = P @ V : A = P (stride 224, K-tail zeros), B = vt (contiguous), K=224 guard-free
__global__ void attn_ctx_kernel(const h16* __restrict__ P, const h16* __restrict__ vt,
                                float* __restrict__ ctx, h16* __restrict__ ctxh) {
  int lane = threadIdx.x & 31;
  int g = lane >> 4;
  int nt = blockIdx.x, mt = blockIdx.y, bh = blockIdx.z;
  int b = bh >> 2, hd = bh & 3;
  const h16* arow = P + ((size_t)bh * Lp + mt * 16 + (lane & 15)) * Lpk + (g << 4);
  const h16* bcol = vt + ((size_t)(bh * DH + nt * 16 + (lane & 15))) * Lpk + (g << 4);
  v8f c = {};
  for (int k0 = 0; k0 < Lpk; k0 += 32) {
    v16h a = cat8(*(const v8h*)(arow + k0), *(const v8h*)(arow + k0 + 8));
    v16h bf = cat8(*(const v8h*)(bcol + k0), *(const v8h*)(bcol + k0 + 8));
    c = __builtin_amdgcn_wmma_f32_16x16x32_f16(false, a, false, bf, (short)0, c, false, false);
  }
  int n = nt * 16 + (lane & 15);
  #pragma unroll
  for (int v = 0; v < 8; ++v) {
    int m = mt * 16 + v + (g << 3);
    if (m < Lseq) {
      size_t o = ((size_t)(b * Lseq + m)) * Hdim + hd * DH + n;
      ctx[o] = c[v]; ctxh[o] = (h16)c[v];
    }
  }
}

// ---------------------------------------------------------------- epilogue pieces
__global__ void add_ln_kernel(const float* __restrict__ x, const float* __restrict__ res,
                              const float* __restrict__ g, const float* __restrict__ bt,
                              float* __restrict__ out, h16* __restrict__ outh) {
  __shared__ float red[Hdim];
  int row = blockIdx.x, t = threadIdx.x;
  size_t off = (size_t)row * Hdim + t;
  float v = x[off] + res[off];
  red[t] = v; __syncthreads();
  for (int st = 128; st > 0; st >>= 1) { if (t < st) red[t] += red[t + st]; __syncthreads(); }
  float mean = red[0] * (1.f / Hdim); __syncthreads();
  float dv = v - mean;
  red[t] = dv * dv; __syncthreads();
  for (int st = 128; st > 0; st >>= 1) { if (t < st) red[t] += red[t + st]; __syncthreads(); }
  float o = dv * rsqrtf(red[0] * (1.f / Hdim) + 1e-6f) * g[t] + bt[t];
  out[off] = o; if (outh) outh[off] = (h16)o;
}

__global__ void select_kernel(const int* __restrict__ gate, const float* __restrict__ mo,
                              const float* __restrict__ ao, float* __restrict__ out,
                              h16* __restrict__ outh) {
  int i = blockIdx.x * blockDim.x + threadIdx.x;
  if (i >= Bsz * Lseq * Hdim) return;
  int b = i / (Lseq * Hdim);
  float v = (gate[b] == 0) ? mo[i] : ao[i];
  out[i] = v; outh[i] = (h16)v;
}

__global__ void logits_kernel(const float* __restrict__ feats, const float* __restrict__ item_emb,
                              const int* __restrict__ pos, const int* __restrict__ neg,
                              float* __restrict__ out) {
  __shared__ float red[Hdim];
  int row = blockIdx.x, t = threadIdx.x;
  float f = feats[(size_t)row * Hdim + t];
  red[t] = f * item_emb[(size_t)pos[row] * Hdim + t]; __syncthreads();
  for (int st = 128; st > 0; st >>= 1) { if (t < st) red[t] += red[t + st]; __syncthreads(); }
  if (t == 0) out[row] = red[0];
  __syncthreads();
  red[t] = f * item_emb[(size_t)neg[row] * Hdim + t]; __syncthreads();
  for (int st = 128; st > 0; st >>= 1) { if (t < st) red[t] += red[t + st]; __syncthreads(); }
  if (t == 0) out[Bsz * Lseq + row] = red[0];
}

// ---------------------------------------------------------------- launch
extern "C" void kernel_launch(void* const* d_in, const int* in_sizes, int n_in,
                              void* d_out, int out_size, void* d_ws, size_t ws_size,
                              hipStream_t stream) {
  (void)in_sizes; (void)n_in; (void)out_size; (void)ws_size;
  const int*   log_seqs   = (const int*)d_in[1];
  const int*   pos_seqs   = (const int*)d_in[2];
  const int*   neg_seqs   = (const int*)d_in[3];
  const float* item_emb   = (const float*)d_in[4];
  const float* pos_emb    = (const float*)d_in[5];
  const float* gate_w     = (const float*)d_in[6];
  const float* gate_b     = (const float*)d_in[7];
  const float* in_proj_w  = (const float*)d_in[8];
  const float* conv_w     = (const float*)d_in[9];
  const float* conv_b     = (const float*)d_in[10];
  const float* x_proj_w   = (const float*)d_in[11];
  const float* dt_w       = (const float*)d_in[12];
  const float* dt_b       = (const float*)d_in[13];
  const float* A_log      = (const float*)d_in[14];
  const float* D_skip     = (const float*)d_in[15];
  const float* out_proj_w = (const float*)d_in[16];
  const float* mha_w      = (const float*)d_in[17];
  const float* mha_b      = (const float*)d_in[18];
  const float* mha_ow     = (const float*)d_in[19];
  const float* mha_ob     = (const float*)d_in[20];
  const float* ln1_g      = (const float*)d_in[21];
  const float* ln1_b      = (const float*)d_in[22];
  const float* ff_w1      = (const float*)d_in[23];
  const float* ff_b1      = (const float*)d_in[24];
  const float* ff_w2      = (const float*)d_in[25];
  const float* ff_b2      = (const float*)d_in[26];
  const float* ln2_g      = (const float*)d_in[27];
  const float* ln2_b      = (const float*)d_in[28];

  char* p = (char*)d_ws;
  auto alloc = [&](size_t bytes) -> void* {
    void* r = (void*)p; p += (bytes + 255) & ~(size_t)255; return r;
  };
  const size_t ML = (size_t)Bsz * Lseq;               // 12800 rows (multiple of 128)

  float* seqs   = (float*)alloc(ML * Hdim * 4);
  h16*   seqs_h = (h16*)  alloc(ML * Hdim * 2);
  h16* inpw_h  = (h16*)alloc((size_t)2 * DIc * Hdim * 2);
  h16* xprw_h  = (h16*)alloc((size_t)(DRr + 2 * Nst) * DIc * 2);
  h16* dtwp_h  = (h16*)alloc((size_t)DIc * 32 * 2);           // zero-padded to K=32
  h16* outw_h  = (h16*)alloc((size_t)Hdim * DIc * 2);
  h16* mhaw_h  = (h16*)alloc((size_t)3 * Hdim * Hdim * 2);
  h16* mhaow_h = (h16*)alloc((size_t)Hdim * Hdim * 2);
  h16* ff1w_h  = (h16*)alloc((size_t)4 * Hdim * Hdim * 2);
  h16* ff2w_h  = (h16*)alloc((size_t)Hdim * 4 * Hdim * 2);
  int*   gate  = (int*)  alloc(Bsz * 4);
  float* xz    = (float*)alloc(ML * 1024 * 4);
  float* xc    = (float*)alloc(ML * DIc * 4);
  h16*   xc_h  = (h16*)  alloc(ML * DIc * 2);
  float* dbl   = (float*)alloc(ML * 80 * 4);
  h16*   dbl_h = (h16*)  alloc(ML * 80 * 2);
  float* delta = (float*)alloc(ML * DIc * 4);
  float* yg    = (float*)alloc(ML * DIc * 4);
  h16*   yg_h  = (h16*)  alloc(ML * DIc * 2);
  float* mo    = (float*)alloc(ML * Hdim * 4);
  h16*   qkv_h = (h16*)  alloc((ML + QKV_PAD_ROWS) * QKVW * 2);  // pad rows zeroed
  float* S     = (float*)alloc((size_t)Bsz * NHEADS * Lp * Lp * 4);
  h16*   P     = (h16*)  alloc((size_t)Bsz * NHEADS * Lp * Lpk * 2);
  h16*   vt    = (h16*)  alloc((size_t)Bsz * NHEADS * DH * Lpk * 2);
  float* ctx   = (float*)alloc(ML * Hdim * 4);
  h16*   ctx_h = (h16*)  alloc(ML * Hdim * 2);
  float* alin  = (float*)alloc(ML * Hdim * 4);
  float* ao    = (float*)alloc(ML * Hdim * 4);
  float* osel  = (float*)alloc(ML * Hdim * 4);
  h16*   osel_h= (h16*)  alloc(ML * Hdim * 2);
  h16*   ffh_h = (h16*)  alloc(ML * 4 * Hdim * 2);
  float* ff2o  = (float*)alloc(ML * Hdim * 4);
  float* feats = (float*)alloc(ML * Hdim * 4);

  auto cvt = [&](const float* s, h16* d, int n) {
    cvt_f16_kernel<<<(n + 255) / 256, 256, 0, stream>>>(s, d, n);
  };
  cvt(in_proj_w,  inpw_h,  2 * DIc * Hdim);
  cvt(x_proj_w,   xprw_h,  (DRr + 2 * Nst) * DIc);
  cvt(out_proj_w, outw_h,  Hdim * DIc);
  cvt(mha_w,      mhaw_h,  3 * Hdim * Hdim);
  cvt(mha_ow,     mhaow_h, Hdim * Hdim);
  cvt(ff_w1,      ff1w_h,  4 * Hdim * Hdim);
  cvt(ff_w2,      ff2w_h,  Hdim * 4 * Hdim);
  dtw_pad_kernel<<<(DIc * 32 + 255) / 256, 256, 0, stream>>>(dt_w, dtwp_h);
  zfill_h16_kernel<<<(QKV_PAD_ROWS * QKVW + 255) / 256, 256, 0, stream>>>(
      qkv_h + ML * QKVW, QKV_PAD_ROWS * QKVW);

  auto gemm = [&](const h16* A, int lda, const h16* W, int ldw, const float* bias,
                  float* C, h16* Ch, int ldc, int M, int N, int K, int act) {
    dim3 g((N + 31) / 32, M / 128);       // N tile per block (uniform), 4 waves split M
    gemm_wmma_kernel<<<g, 128, 0, stream>>>(A, lda, W, ldw, bias, C, Ch, ldc, N, K, act);
  };

  int nBLH = (int)(ML * Hdim), nBLD = (int)(ML * DIc);
  embed_kernel<<<(nBLH + 255) / 256, 256, 0, stream>>>(log_seqs, item_emb, pos_emb, seqs, seqs_h);
  gate_kernel<<<Bsz, Hdim, 0, stream>>>(seqs, gate_w, gate_b, gate);

  // ---- Mamba branch
  gemm(seqs_h, Hdim, inpw_h, Hdim, nullptr, xz, nullptr, 1024, (int)ML, 1024, Hdim, 0);
  conv_silu_kernel<<<(nBLD + 255) / 256, 256, 0, stream>>>(xz, conv_w, conv_b, xc, xc_h);
  gemm(xc_h, DIc, xprw_h, DIc, nullptr, dbl, dbl_h, 80, (int)ML, 80, DIc, 0);
  gemm(dbl_h, 80, dtwp_h, 32, dt_b, delta, nullptr, DIc, (int)ML, DIc, 32, 3); // softplus, padded K
  scan_kernel<<<(Bsz * DIc + 255) / 256, 256, 0, stream>>>(delta, dbl, xc, xz, A_log, D_skip, yg, yg_h);
  gemm(yg_h, DIc, outw_h, DIc, nullptr, mo, nullptr, Hdim, (int)ML, Hdim, DIc, 0);

  // ---- Attention branch
  gemm(seqs_h, Hdim, mhaw_h, Hdim, mha_b, nullptr, qkv_h, QKVW, (int)ML, QKVW, Hdim, 0);
  vtrans_kernel<<<(Bsz * NHEADS * DH * Lpk + 255) / 256, 256, 0, stream>>>(qkv_h, vt);
  attn_scores_kernel<<<dim3(13, 13, Bsz * NHEADS), 32, 0, stream>>>(qkv_h, S);
  softmax_kernel<<<Bsz * NHEADS * Lseq, 256, 0, stream>>>(S, P);
  attn_ctx_kernel<<<dim3(4, 13, Bsz * NHEADS), 32, 0, stream>>>(P, vt, ctx, ctx_h);
  gemm(ctx_h, Hdim, mhaow_h, Hdim, mha_ob, alin, nullptr, Hdim, (int)ML, Hdim, Hdim, 0);
  add_ln_kernel<<<(int)ML, Hdim, 0, stream>>>(alin, seqs, ln1_g, ln1_b, ao, nullptr);

  // ---- gate select + FFN + LN2 + logits
  select_kernel<<<(nBLH + 255) / 256, 256, 0, stream>>>(gate, mo, ao, osel, osel_h);
  gemm(osel_h, Hdim, ff1w_h, Hdim, ff_b1, nullptr, ffh_h, 1024, (int)ML, 1024, Hdim, 2); // relu
  gemm(ffh_h, 1024, ff2w_h, 1024, ff_b2, ff2o, nullptr, Hdim, (int)ML, Hdim, 1024, 0);
  add_ln_kernel<<<(int)ML, Hdim, 0, stream>>>(ff2o, osel, ln2_g, ln2_b, feats, nullptr);
  logits_kernel<<<(int)ML, Hdim, 0, stream>>>(feats, item_emb, pos_seqs, neg_seqs, (float*)d_out);
}
